// RopeAttentionModel_35296041238714
// MI455X (gfx1250) — matrix operand order repaired and checked by reading, not compiled, not run on hardware
//
#include <hip/hip_runtime.h>
#include <hip/hip_bf16.h>
#include <math.h>

#define S_LEN  2048
#define HD     128
#define NQH    32
#define NKVH   8
#define HIDDEN 4096

typedef __attribute__((ext_vector_type(16))) _Float16 v16h;
typedef __attribute__((ext_vector_type(8)))  _Float16 v8h;
typedef __attribute__((ext_vector_type(8)))  float    v8f;

// ---------------- utility: zero a float buffer ----------------
__global__ void zero_f32_kernel(float* __restrict__ p, int n) {
    int i = blockIdx.x * 256 + threadIdx.x;
    if (i < n) p[i] = 0.0f;
}

// ---------------- column sums of a [4096, cols] matrix ----------------
// grid: (cols/256, 16); each block sums 256 rows for 256 columns, atomicAdd.
__global__ __launch_bounds__(256) void colsum_kernel(const float* __restrict__ W,
                                                     float* __restrict__ out,
                                                     int cols) {
    int col = blockIdx.x * 256 + threadIdx.x;
    int r0  = blockIdx.y * 256;
    const float* p = W + (size_t)r0 * cols + col;
    float s = 0.0f;
#pragma unroll 8
    for (int i = 0; i < 256; ++i) s += p[(size_t)i * cols];
    atomicAdd(&out[col], s);
}

// ---------------- M[h, j] = sum_d cv[kv(h), d] * Wo[h*128+d, j] ----------------
// grid: (4096/256, 32)
__global__ __launch_bounds__(256) void mproj_kernel(const float* __restrict__ cv,
                                                    const float* __restrict__ Wo,
                                                    float* __restrict__ M) {
    int j = blockIdx.x * 256 + threadIdx.x;
    int h = blockIdx.y;
    const float* cvr = cv + (size_t)(h >> 2) * HD;
    const float* wo  = Wo + (size_t)h * HD * HIDDEN + j;
    float s = 0.0f;
#pragma unroll 4
    for (int d = 0; d < HD; ++d) s += cvr[d] * wo[(size_t)d * HIDDEN];
    M[(size_t)h * HIDDEN + j] = s;
}

// ---------------- RoPE'd rank-1 q/k in f16, head-major; also ids -> f32 ----------------
// qr: [NQH][S][HD], kr: [NKVH][S][HD].  grid: (S), block: 128 (d = tid)
__global__ __launch_bounds__(128) void rope_kernel(const int* __restrict__ ids,
                                                   const int* __restrict__ pos,
                                                   const float* __restrict__ cq,
                                                   const float* __restrict__ ck,
                                                   _Float16* __restrict__ qr,
                                                   _Float16* __restrict__ kr,
                                                   float* __restrict__ idsf) {
    const int s = blockIdx.x;
    const int d = threadIdx.x;
    const float p   = (float)pos[s];
    const int   fi  = d & 63;
    // inv_freq = 10000^(-fi/64)
    const float inv = expf(-(float)fi * (9.210340371976184f / 64.0f));
    const float ang = p * inv;
    const float cs  = cosf(ang);
    const float sn  = sinf(ang);
    const float idv = (float)ids[s];
    if (d == 0) idsf[s] = idv;
#pragma unroll 1
    for (int h = 0; h < NQH; ++h) {
        const float a = cq[h * HD + d];
        const float b = (d < 64) ? -cq[h * HD + d + 64] : cq[h * HD + d - 64];
        qr[((size_t)h * S_LEN + s) * HD + d] = (_Float16)(idv * (a * cs + b * sn));
    }
#pragma unroll 1
    for (int h = 0; h < NKVH; ++h) {
        const float a = ck[h * HD + d];
        const float b = (d < 64) ? -ck[h * HD + d + 64] : ck[h * HD + d - 64];
        kr[((size_t)h * S_LEN + s) * HD + d] = (_Float16)(idv * (a * cs + b * sn));
    }
}

// ---------------- flash attention with scalar value id[t] ----------------
// grid: (NQH, S/128), block: 128 (4 waves, one 32-row q-tile per wave)
// w[s, h] = sum_t softmax(q_r[s]·k_r[t]/sqrt(128), causal) * id[t]
//
// Two A fragment sets + two independent D chains share each B tile (halves
// L2 traffic per WMMA). Per-lane online softmax state in the k-loop; the
// 16-lane (m, l, acc) merge happens once at the end.
__global__ __launch_bounds__(128) void attn_kernel(const _Float16* __restrict__ qr,
                                                   const _Float16* __restrict__ kr,
                                                   const float* __restrict__ idsf,
                                                   float* __restrict__ w) {
    const int head = blockIdx.x;
    const int wave = threadIdx.x >> 5;
    const int lane = threadIdx.x & 31;
    const int n    = lane & 15;
    const int half = lane >> 4;
    const int s0   = (blockIdx.y * 4 + wave) * 32;   // 32 q-rows per wave

    const _Float16* kh = kr + (size_t)(head >> 2) * S_LEN * HD;

    // A fragments: rows [s0, s0+16) in a0, rows [s0+16, s0+32) in a1.
    // 16-bit A 16x32: lane(m,half): j -> K=half*16+j (matches B)
    v16h a0[4], a1[4];
    {
        const _Float16* q0 = qr + ((size_t)head * S_LEN + (s0 + n)) * HD;
        const _Float16* q1 = q0 + (size_t)16 * HD;
#pragma unroll
        for (int c = 0; c < 4; ++c) {
            v8h lo = *(const v8h*)(q0 + c * 32 + half * 16);
            v8h hi = *(const v8h*)(q0 + c * 32 + half * 16 + 8);
            a0[c] = __builtin_shufflevector(lo, hi, 0, 1, 2, 3, 4, 5, 6, 7,
                                                    8, 9, 10, 11, 12, 13, 14, 15);
            lo = *(const v8h*)(q1 + c * 32 + half * 16);
            hi = *(const v8h*)(q1 + c * 32 + half * 16 + 8);
            a1[c] = __builtin_shufflevector(lo, hi, 0, 1, 2, 3, 4, 5, 6, 7,
                                                    8, 9, 10, 11, 12, 13, 14, 15);
        }
    }

    float m_ln[16], l_ln[16], a_ln[16];
#pragma unroll
    for (int r = 0; r < 16; ++r) { m_ln[r] = -1e30f; l_ln[r] = 0.0f; a_ln[r] = 0.0f; }

    const float sc = 0.08838834764831845f; // 1/sqrt(128)

    // one 16-key tile: 8 WMMA (two chains) + per-lane softmax update
    auto tile = [&](int t0, bool mask_tile) {
        // 16-bit B 32x16 (this code's contiguous k order, same as A): lane(n,half): j -> K = half*16+j, column n
        const _Float16* krow = kh + (size_t)(t0 + n) * HD;
        const v16h b0 = *(const v16h*)(krow + 0 * 32 + half * 16);
        const v16h b1 = *(const v16h*)(krow + 1 * 32 + half * 16);
        const v16h b2 = *(const v16h*)(krow + 2 * 32 + half * 16);
        const v16h b3 = *(const v16h*)(krow + 3 * 32 + half * 16);

        v8f d0 = {}, d1 = {};
        d0 = __builtin_amdgcn_wmma_f32_16x16x32_f16(false, a0[0], false, b0, (short)0, d0, false, false);
        d1 = __builtin_amdgcn_wmma_f32_16x16x32_f16(false, a1[0], false, b0, (short)0, d1, false, false);
        d0 = __builtin_amdgcn_wmma_f32_16x16x32_f16(false, a0[1], false, b1, (short)0, d0, false, false);
        d1 = __builtin_amdgcn_wmma_f32_16x16x32_f16(false, a1[1], false, b1, (short)0, d1, false, false);
        d0 = __builtin_amdgcn_wmma_f32_16x16x32_f16(false, a0[2], false, b2, (short)0, d0, false, false);
        d1 = __builtin_amdgcn_wmma_f32_16x16x32_f16(false, a1[2], false, b2, (short)0, d1, false, false);
        d0 = __builtin_amdgcn_wmma_f32_16x16x32_f16(false, a0[3], false, b3, (short)0, d0, false, false);
        d1 = __builtin_amdgcn_wmma_f32_16x16x32_f16(false, a1[3], false, b3, (short)0, d1, false, false);

        const int   tcol = t0 + n;
        const float idt  = idsf[tcol];

#pragma unroll
        for (int g = 0; g < 2; ++g) {
#pragma unroll
            for (int r = 0; r < 8; ++r) {
                const int   idx = g * 8 + r;
                const float sv  = (g == 0 ? d0[r] : d1[r]) * sc; // D VGPR r -> row half*8+r
                bool valid = true;
                if (mask_tile) valid = (tcol <= s0 + g * 16 + half * 8 + r);
                const float svm = valid ? sv : -1e30f;
                const float nm  = fmaxf(m_ln[idx], svm);
                const float e   = valid ? __expf(sv - nm) : 0.0f;
                const float rs  = __expf(m_ln[idx] - nm); // m=-1e30 & nm finite -> 0
                l_ln[idx] = l_ln[idx] * rs + e;
                a_ln[idx] = a_ln[idx] * rs + e * idt;
                m_ln[idx] = nm;
            }
        }
    };

    for (int t0 = 0; t0 < s0; t0 += 16) tile(t0, false); // fully-causal tiles
    tile(s0, true);                                      // diagonal tiles
    tile(s0 + 16, true);

    // merge (m, l, acc) across the 16 key-column lanes (once per q-tile)
#pragma unroll
    for (int r = 0; r < 16; ++r) {
#pragma unroll
        for (int off = 1; off < 16; off <<= 1) {
            const float mo = __shfl_xor(m_ln[r], off, 16);
            const float lo = __shfl_xor(l_ln[r], off, 16);
            const float ao = __shfl_xor(a_ln[r], off, 16);
            const float nm = fmaxf(m_ln[r], mo);
            const float s1 = __expf(m_ln[r] - nm);   // both -1e30 -> 1, but l==0
            const float s2 = __expf(mo - nm);
            l_ln[r] = l_ln[r] * s1 + lo * s2;
            a_ln[r] = a_ln[r] * s1 + ao * s2;
            m_ln[r] = nm;
        }
    }

    if (n == 0) {
#pragma unroll
        for (int r = 0; r < 16; ++r) {
            const int row = s0 + (r >> 3) * 16 + half * 8 + (r & 7);
            w[(size_t)row * NQH + head] = a_ln[r] / l_ln[r];
        }
    }
}

// ---------------- out[s, j] = sum_h w[s,h] * M[h,j] ----------------
// grid: (4096/256, S)
__global__ __launch_bounds__(256) void out_kernel(const float* __restrict__ w,
                                                  const float* __restrict__ M,
                                                  float* __restrict__ out) {
    int j = blockIdx.x * 256 + threadIdx.x;
    int s = blockIdx.y;
    const float* wr = w + (size_t)s * NQH;   // uniform per block -> scalar loads
    float sum = 0.0f;
#pragma unroll
    for (int h = 0; h < NQH; ++h) sum += wr[h] * M[(size_t)h * HIDDEN + j];
    out[(size_t)s * HIDDEN + j] = sum;
}

extern "C" void kernel_launch(void* const* d_in, const int* in_sizes, int n_in,
                              void* d_out, int out_size, void* d_ws, size_t ws_size,
                              hipStream_t stream) {
    (void)in_sizes; (void)n_in; (void)out_size; (void)ws_size;
    const int*   ids = (const int*)d_in[0];
    const int*   pos = (const int*)d_in[1];
    const float* Wq  = (const float*)d_in[2];
    const float* Wk  = (const float*)d_in[3];
    const float* Wv  = (const float*)d_in[4];
    const float* Wo  = (const float*)d_in[5];
    float* out = (float*)d_out;

    // workspace layout (all offsets 256B aligned), total ~21 MB
    char* ws = (char*)d_ws;
    float*    cq   = (float*)(ws + 0);                             // 4096 f32
    float*    ck   = (float*)(ws + 16384);                         // 1024 f32
    float*    cv   = (float*)(ws + 20480);                         // 1024 f32
    float*    M    = (float*)(ws + 24576);                         // 32*4096 f32
    float*    w    = (float*)(ws + 24576 + 524288);                // 2048*32 f32
    _Float16* qr   = (_Float16*)(ws + 24576 + 524288 + 262144);    // 32*2048*128 f16
    _Float16* kr   = (_Float16*)((char*)qr + (size_t)NQH * S_LEN * HD * 2);  // 8*2048*128 f16
    float*    idsf = (float*)((char*)kr + (size_t)NKVH * S_LEN * HD * 2);    // 2048 f32

    // 1) zero cq|ck|cv (contiguous 6144 floats)
    zero_f32_kernel<<<dim3(24), 256, 0, stream>>>(cq, 6144);

    // 2) rank-1 collapse of the QKV projections: column sums of W
    colsum_kernel<<<dim3(HIDDEN / 256, 16), 256, 0, stream>>>(Wq, cq, HIDDEN);
    colsum_kernel<<<dim3((NKVH * HD) / 256, 16), 256, 0, stream>>>(Wk, ck, NKVH * HD);
    colsum_kernel<<<dim3((NKVH * HD) / 256, 16), 256, 0, stream>>>(Wv, cv, NKVH * HD);

    // 3) fold V and Wo: M[h,:] = cv[kv(h),:] @ Wo[h-block,:]
    mproj_kernel<<<dim3(HIDDEN / 256, NQH), 256, 0, stream>>>(cv, Wo, M);

    // 4) RoPE'd q/k in f16 (head-major) for the WMMA score GEMM
    rope_kernel<<<dim3(S_LEN), 128, 0, stream>>>(ids, pos, cq, ck, qr, kr, idsf);

    // 5) causal flash attention with scalar value stream -> w[s,h]
    attn_kernel<<<dim3(NQH, S_LEN / 128), 128, 0, stream>>>(qr, kr, idsf, w);

    // 6) epilogue GEMM: out = w @ M
    out_kernel<<<dim3(HIDDEN / 256, S_LEN), 256, 0, stream>>>(w, M, out);
}
